// FHN_LSTM_41412074668788
// MI455X (gfx1250) — matrix operand order repaired and checked by reading, not compiled, not run on hardware
//
#include <hip/hip_runtime.h>

// FHN-LSTM recurrent solver for MI455X (gfx1250, wave32, WMMA).
// Persistent 16-block kernel, 2 grid barriers per time step, f16 WMMA math,
// rcp-based activations to shrink the per-step VALU tail.

#define TSTEPS 1000
#define BATCH  64
#define HID    512
#define GBLK   16    // persistent blocks (each on its own WGP)
#define TPB    256   // 8 wave32s per block -> 128 waves total = 128 tile jobs

typedef __attribute__((ext_vector_type(16))) _Float16 v16h;
typedef __attribute__((ext_vector_type(8)))  _Float16 v8h;
typedef __attribute__((ext_vector_type(8)))  float    v8f;

// ---------------- workspace layout (bytes) ----------------
static constexpr size_t OFF_W0   = 0;                                   // 2048*512 f16
static constexpr size_t OFF_W1   = OFF_W0  + (size_t)2048 * 512 * 2;    // 2048*512 f16
static constexpr size_t OFF_BS0  = OFF_W1  + (size_t)2048 * 512 * 2;    // 2048 f32
static constexpr size_t OFF_BS1  = OFF_BS0 + (size_t)2048 * 4;          // 2048 f32
static constexpr size_t OFF_H    = OFF_BS1 + (size_t)2048 * 4;          // 64*512 f16
static constexpr size_t OFF_HA   = OFF_H   + (size_t)BATCH * HID * 2;   // 64*512 f16
static constexpr size_t OFF_SYNC = OFF_HA  + (size_t)BATCH * HID * 2;   // sync words

// ---------------- helpers ----------------
// v_rcp_f32-based fast activations (avoid IEEE div_scale/div_fma expansion).
__device__ __forceinline__ float fast_rcp(float x) {
    return __builtin_amdgcn_rcpf(x);
}
__device__ __forceinline__ float sigmoidf_(float x) {
    return fast_rcp(1.0f + __expf(-x));                 // v_exp + v_add + v_rcp
}
__device__ __forceinline__ float tanhf_(float x) {
    return 1.0f - 2.0f * fast_rcp(__expf(2.0f * x) + 1.0f);
}

// A fragment (16x32 f16, row-major source, stride HID):
// lane L: m = m0+(L&15); halfs 0..15 -> K = k0+16*(L>>4)+{0..15}.
__device__ __forceinline__ v16h load_a_frag(const _Float16* __restrict__ A,
                                            int m0, int k0, int lane) {
    const int m  = m0 + (lane & 15);
    const int kb = k0 + 16 * (lane >> 4);
    const _Float16* p = A + (size_t)m * HID + kb;
    return *(const v16h*)(p);
}

// B fragment (32x16 f16). B = W^T with W row-major [2048,512], so
// lane L: column n = n0+(L&15) == W row, 16 contiguous halfs at k0+16*(L>>4).
__device__ __forceinline__ v16h load_b_frag(const _Float16* __restrict__ W,
                                            int n0, int k0, int lane) {
    const int n  = n0 + (lane & 15);
    const int kb = k0 + 16 * (lane >> 4);
    return *(const v16h*)(W + (size_t)n * HID + kb);
}

// One degenerate-LSTM layer for a 16x16 output tile (m0,j0):
// gates(i,g,o) = Hin * W^T + bs ; Hout = sigm(o)*tanh(sigm(i)*tanh(g))
__device__ __forceinline__ void lstm_layer(const _Float16* __restrict__ Hin,
                                           _Float16* __restrict__ Hout,
                                           const _Float16* __restrict__ W,
                                           const float* __restrict__ bs,
                                           int m0, int j0, int lane) {
    v8f acc_i = {}; v8f acc_g = {}; v8f acc_o = {};
#pragma unroll 2
    for (int kt = 0; kt < HID / 32; ++kt) {
        const int k0 = kt * 32;
        v16h a  = load_a_frag(Hin, m0, k0, lane);
        v16h bi = load_b_frag(W, j0,        k0, lane);   // i gate rows [0,512)
        v16h bg = load_b_frag(W, 1024 + j0, k0, lane);   // g gate rows [1024,1536)
        v16h bo = load_b_frag(W, 1536 + j0, k0, lane);   // o gate rows [1536,2048)
        acc_i = __builtin_amdgcn_wmma_f32_16x16x32_f16(false, a, false, bi,
                                                       (short)0, acc_i, false, false);
        acc_g = __builtin_amdgcn_wmma_f32_16x16x32_f16(false, a, false, bg,
                                                       (short)0, acc_g, false, false);
        acc_o = __builtin_amdgcn_wmma_f32_16x16x32_f16(false, a, false, bo,
                                                       (short)0, acc_o, false, false);
    }
    // D layout: lane L -> n = j0+(L&15); VGPR r -> m = m0 + r + 8*(L>>4)
    const int jc    = j0 + (lane & 15);
    const int mbase = m0 + 8 * (lane >> 4);
    const float bib = bs[jc];
    const float bgb = bs[1024 + jc];
    const float bob = bs[1536 + jc];
#pragma unroll
    for (int r = 0; r < 8; ++r) {
        float iv = acc_i[r] + bib;
        float gv = acc_g[r] + bgb;
        float ov = acc_o[r] + bob;
        float c  = sigmoidf_(iv) * tanhf_(gv);
        float h  = sigmoidf_(ov) * tanhf_(c);
        Hout[(size_t)(mbase + r) * HID + jc] = (_Float16)h;
    }
}

// Generation-counter grid barrier (device scope).
__device__ __forceinline__ void grid_barrier(unsigned* counter, unsigned* gen,
                                             unsigned nblk) {
    __builtin_amdgcn_fence(__ATOMIC_RELEASE, "agent");  // publish prior stores
    __syncthreads();
    if (threadIdx.x == 0) {
        unsigned g = __hip_atomic_load(gen, __ATOMIC_RELAXED, __HIP_MEMORY_SCOPE_AGENT);
        unsigned a = __hip_atomic_fetch_add(counter, 1u, __ATOMIC_ACQ_REL,
                                            __HIP_MEMORY_SCOPE_AGENT);
        if (a == nblk - 1) {
            __hip_atomic_store(counter, 0u, __ATOMIC_RELAXED, __HIP_MEMORY_SCOPE_AGENT);
            __hip_atomic_store(gen, g + 1, __ATOMIC_RELEASE, __HIP_MEMORY_SCOPE_AGENT);
        } else {
            while (__hip_atomic_load(gen, __ATOMIC_ACQUIRE,
                                     __HIP_MEMORY_SCOPE_AGENT) == g) {
                __builtin_amdgcn_s_sleep(8);
            }
        }
    }
    __syncthreads();
    __builtin_amdgcn_fence(__ATOMIC_ACQUIRE, "agent"); // invalidate stale L0 lines
}

// ---------------- prep kernels ----------------
__global__ void k_f32_to_f16(const float* __restrict__ s, _Float16* __restrict__ d, int n) {
    int i = blockIdx.x * blockDim.x + threadIdx.x;
    if (i < n) d[i] = (_Float16)s[i];
}

__global__ void k_bias_sum(const float* __restrict__ a, const float* __restrict__ b,
                           float* __restrict__ o, int n) {
    int i = blockIdx.x * blockDim.x + threadIdx.x;
    if (i < n) o[i] = a[i] + b[i];
}

// x0[b][j] = u0[b]*W_in[j,0] + w0[b]*W_in[j,1] + K[b]*W_in[j,2] + b_in[j]
__global__ void k_init_x(const float* __restrict__ u0, const float* __restrict__ w0,
                         const float* __restrict__ K,  const float* __restrict__ W_in,
                         const float* __restrict__ b_in, _Float16* __restrict__ H) {
    int i = blockIdx.x * blockDim.x + threadIdx.x;
    if (i < BATCH * HID) {
        int b = i >> 9, j = i & (HID - 1);
        float v = u0[b] * W_in[j * 3 + 0] + w0[b] * W_in[j * 3 + 1] +
                  K[b]  * W_in[j * 3 + 2] + b_in[j];
        H[i] = (_Float16)v;
    }
}

// ---------------- persistent recurrence kernel ----------------
__global__ __launch_bounds__(TPB) void k_fhn_persist(
    const _Float16* __restrict__ W0, const _Float16* __restrict__ W1,
    const float* __restrict__ bs0,   const float* __restrict__ bs1,
    const float* __restrict__ Wu,    const float* __restrict__ Ww,
    const float* __restrict__ bu,    const float* __restrict__ bw,
    _Float16* __restrict__ H, _Float16* __restrict__ HA,
    float* __restrict__ out, unsigned* __restrict__ sync) {
    __shared__ float red_u[TPB], red_w[TPB];
    const int lane = threadIdx.x & 31;
    const int wave = blockIdx.x * (TPB >> 5) + (threadIdx.x >> 5); // 0..127
    const int m0   = (wave >> 5) * 16;   // 4 M-tiles (batch)
    const int j0   = (wave & 31) * 16;   // 32 j-tiles (hidden)

    for (int t = 0; t < TSTEPS; ++t) {
        lstm_layer(H, HA, W0, bs0, m0, j0, lane);     // layer 0: H -> HA
        grid_barrier(sync, sync + 1, GBLK);
        lstm_layer(HA, H, W1, bs1, m0, j0, lane);     // layer 1: HA -> H (carry)
        grid_barrier(sync, sync + 1, GBLK);

        if (blockIdx.x == 0) {                         // u,w readout from H(t)
            const int b    = threadIdx.x >> 2;         // 0..63
            const int part = threadIdx.x & 3;
            const _Float16* hr = H + (size_t)b * HID + part * 128;
            const float* wu = Wu + part * 128;
            const float* ww = Ww + part * 128;
            float su = 0.f, sw = 0.f;
            for (int j = 0; j < 128; j += 8) {         // vectorized b128 f16 loads
                v8h hv = *(const v8h*)(hr + j);
#pragma unroll
                for (int e = 0; e < 8; ++e) {
                    float h = (float)hv[e];
                    su += h * wu[j + e];
                    sw += h * ww[j + e];
                }
            }
            red_u[threadIdx.x] = su; red_w[threadIdx.x] = sw;
            __syncthreads();
            if (part == 0) {
                int x = threadIdx.x;
                float uo = red_u[x] + red_u[x+1] + red_u[x+2] + red_u[x+3] + bu[0];
                float wo = red_w[x] + red_w[x+1] + red_w[x+2] + red_w[x+3] + bw[0];
                out[b * TSTEPS + t] = uo;                       // us [B,T]
                out[BATCH * TSTEPS + b * TSTEPS + t] = wo;      // ws [B,T]
            }
            // red_* reuse is fenced by the grid_barrier __syncthreads next iter
        }
    }
}

extern "C" void kernel_launch(void* const* d_in, const int* in_sizes, int n_in,
                              void* d_out, int out_size, void* d_ws, size_t ws_size,
                              hipStream_t stream) {
    // setup_inputs order:
    const float* u0    = (const float*)d_in[0];
    const float* w0    = (const float*)d_in[1];
    const float* K     = (const float*)d_in[2];
    const float* W_in  = (const float*)d_in[3];
    const float* b_in  = (const float*)d_in[4];
    const float* W_ih0 = (const float*)d_in[5];
    const float* b_ih0 = (const float*)d_in[7];
    const float* b_hh0 = (const float*)d_in[8];
    const float* W_ih1 = (const float*)d_in[9];
    const float* b_ih1 = (const float*)d_in[11];
    const float* b_hh1 = (const float*)d_in[12];
    const float* W_u   = (const float*)d_in[13];
    const float* b_u   = (const float*)d_in[14];
    const float* W_w   = (const float*)d_in[15];
    const float* b_w   = (const float*)d_in[16];

    char* ws = (char*)d_ws;
    _Float16* W0h = (_Float16*)(ws + OFF_W0);
    _Float16* W1h = (_Float16*)(ws + OFF_W1);
    float*    bs0 = (float*)(ws + OFF_BS0);
    float*    bs1 = (float*)(ws + OFF_BS1);
    _Float16* H   = (_Float16*)(ws + OFF_H);
    _Float16* HA  = (_Float16*)(ws + OFF_HA);
    unsigned* syn = (unsigned*)(ws + OFF_SYNC);
    float*    out = (float*)d_out;

    // zero grid-barrier state every launch (graph-capturable memset node)
    hipMemsetAsync(syn, 0, 64, stream);

    const int NW = 2048 * 512;
    k_f32_to_f16<<<(NW + 255) / 256, 256, 0, stream>>>(W_ih0, W0h, NW);
    k_f32_to_f16<<<(NW + 255) / 256, 256, 0, stream>>>(W_ih1, W1h, NW);
    k_bias_sum<<<(2048 + 255) / 256, 256, 0, stream>>>(b_ih0, b_hh0, bs0, 2048);
    k_bias_sum<<<(2048 + 255) / 256, 256, 0, stream>>>(b_ih1, b_hh1, bs1, 2048);
    k_init_x<<<(BATCH * HID + 255) / 256, 256, 0, stream>>>(u0, w0, K, W_in, b_in, H);

    k_fhn_persist<<<GBLK, TPB, 0, stream>>>(W0h, W1h, bs0, bs1, W_u, W_w,
                                            b_u, b_w, H, HA, out, syn);
}
